// DeltaDecoder_82712480186639
// MI455X (gfx1250) — matrix operand order repaired and checked by reading, not compiled, not run on hardware
//
#include <hip/hip_runtime.h>
#include <math.h>

typedef _Float16 f16;
typedef __attribute__((ext_vector_type(16))) _Float16 v16h;
typedef __attribute__((ext_vector_type(8)))  _Float16 v8h;
typedef __attribute__((ext_vector_type(8)))  float    v8f;
typedef __attribute__((ext_vector_type(4)))  float    v4f;
typedef __attribute__((ext_vector_type(2)))  float    v2f;

#define BATCH 16384
#define CTX 256
#define HID 128
#define NG 384           // 3*HID
#define TSTEPS 48
#define DTIME (1.0f/30.0f)

static __device__ __forceinline__ v16h cat16(v8h lo, v8h hi) {
  return __builtin_shufflevector(lo, hi, 0,1,2,3,4,5,6,7,8,9,10,11,12,13,14,15);
}

static __device__ __forceinline__ v8f wmma_f16(v16h a, v16h b, v8f c) {
  return __builtin_amdgcn_wmma_f32_16x16x32_f16(false, a, false, b, (short)0, c, false, false);
}

// load a pre-swizzled B fragment from LDS: [tile][lane][16 f16]
static __device__ __forceinline__ v16h ldsB(const _Float16* base, int tile, int ln) {
  const _Float16* p = base + tile * 512 + ln * 16;
  v8h lo = *(const v8h*)(p);
  v8h hi = *(const v8h*)(p + 8);
  return cat16(lo, hi);
}

// scheduling fence: nothing may be moved across (bounds VGPR live range)
static __device__ __forceinline__ void sched_fence() {
  __builtin_amdgcn_sched_barrier(0);
}

// branch-free fast activations (correct saturation at +/-inf)
static __device__ __forceinline__ float sigm(float x) {
  return __frcp_rn(1.0f + __expf(-x));
}
static __device__ __forceinline__ float tanh_fast(float x) {
  return 1.0f - 2.0f * __frcp_rn(__expf(2.0f * x) + 1.0f);
}

__global__ __launch_bounds__(64)
void delta_decoder_gru_kernel(const float* __restrict__ ctx,
                              const float* __restrict__ lv,
                              const float* __restrict__ W_init,
                              const float* __restrict__ b_init,
                              const float* __restrict__ W_ih,
                              const float* __restrict__ b_ih,
                              const float* __restrict__ W_hh,
                              const float* __restrict__ b_hh,
                              const float* __restrict__ W1,
                              const float* __restrict__ b1,
                              const float* __restrict__ W2,
                              const float* __restrict__ b2,
                              float* __restrict__ out) {
  // ---- shared memory (~153 KB) ----
  __shared__ __align__(16) _Float16 sWhh[24 * 4 * 512];   // 96 KB, B-layout
  __shared__ __align__(16) _Float16 sW1 [4 * 4 * 512];    // 16 KB, B-layout
  __shared__ __align__(16) _Float16 sW2 [2 * 512];        //  2 KB, B-layout (N padded to 16)
  __shared__ __align__(16) float    sWd [NG * 2];         // W_ih[:, 0:2] as float2[j]
  __shared__ __align__(16) _Float16 sH  [2][16 * 128];    // per-wave h (f16 mirror)
  __shared__ __align__(16) _Float16 sGi [2][24 * 256];    // per-wave gi_ctx (+folded biases)
  __shared__ __align__(16) _Float16 sHid[2][16 * 64];
  __shared__ __align__(16) float    sDelta[2][32];        // [m][2]

  const int tid = threadIdx.x;

  // ---- stage 1: convert recurrent weights into WMMA B-layout in LDS ----
  for (int idx = tid; idx < 24 * 4 * 512; idx += 64) {
    int tile = idx >> 9, w = idx & 511, ln = w >> 4, i = w & 15;
    int nt = tile >> 2, kt = tile & 3;
    int n = nt * 16 + (ln & 15);
    int k = kt * 32 + (ln >> 4) * 16 + i;
    sWhh[idx] = (f16)W_hh[n * 128 + k];
  }
  for (int idx = tid; idx < 4 * 4 * 512; idx += 64) {
    int tile = idx >> 9, w = idx & 511, ln = w >> 4, i = w & 15;
    int nt = tile >> 2, kt = tile & 3;
    int n = nt * 16 + (ln & 15);
    int k = kt * 32 + (ln >> 4) * 16 + i;
    sW1[idx] = (f16)W1[n * 128 + k];
  }
  for (int idx = tid; idx < 2 * 512; idx += 64) {
    int kt = idx >> 9, w = idx & 511, ln = w >> 4, i = w & 15;
    int n = ln & 15;
    int k = kt * 32 + (ln >> 4) * 16 + i;
    sW2[idx] = (n < 2) ? (f16)W2[n * 64 + k] : (f16)0.0f;
  }
  for (int idx = tid; idx < NG * 2; idx += 64) {
    int j = idx >> 1, c = idx & 1;
    sWd[idx] = W_ih[j * 258 + c];
  }
  __syncthreads();

  // ---- per-wave identity ----
  const int wv  = tid >> 5;
  const int ln  = tid & 31;
  const int col = ln & 15;     // N / row-in-A index
  const int hi  = ln >> 4;     // half-wave index
  const int rowBase = (blockIdx.x * 2 + wv) * 16;

  // delta0 = last_velocity * dt
  if (ln < 16) {
    sDelta[wv][ln * 2 + 0] = lv[(rowBase + ln) * 2 + 0] * DTIME;
    sDelta[wv][ln * 2 + 1] = lv[(rowBase + ln) * 2 + 1] * DTIME;
  }

  // ---- stage 2: context A-fragments (8 k-tiles of K=32), float4 loads ----
  v16h actx[8];
  {
    const float* crow = ctx + (size_t)(rowBase + col) * CTX;
#pragma unroll
    for (int kt = 0; kt < 8; kt++) {
      const v4f* p0 = (const v4f*)(crow + kt * 32 + hi * 16);
      const v4f* p1 = (const v4f*)(crow + kt * 32 + hi * 16 + 8);
      v4f q0 = p0[0], q1 = p0[1], q2 = p1[0], q3 = p1[1];
      v16h a;
#pragma unroll
      for (int e = 0; e < 4; e++) {
        a[e]      = (f16)q0[e];
        a[4 + e]  = (f16)q1[e];
        a[8 + e]  = (f16)q2[e];
        a[12 + e] = (f16)q3[e];
      }
      actx[kt] = a;
    }
  }

  // ---- h0 = tanh(context @ W_init.T + b_init) ----
  v8f hst[8];
#pragma unroll
  for (int c = 0; c < 8; c++) {
    v8f acc = {};
    const float* wrow = W_init + (size_t)(c * 16 + col) * CTX;
#pragma unroll
    for (int kt = 0; kt < 8; kt++) {
      const v4f* p = (const v4f*)(wrow + kt * 32 + hi * 16);  // 16B aligned
      v16h b;
#pragma unroll
      for (int u = 0; u < 4; u++) {
        v4f q = p[u];
#pragma unroll
        for (int e = 0; e < 4; e++) b[u * 4 + e] = (f16)q[e];
      }
      acc = wmma_f16(actx[kt], b, acc);
    }
    float bias = b_init[c * 16 + col];
    v8f hv;
#pragma unroll
    for (int r = 0; r < 8; r++) {
      float v = tanh_fast(acc[r] + bias);
      hv[r] = v;
      sH[wv][(hi * 8 + r) * 128 + c * 16 + col] = (f16)v;
    }
    hst[c] = hv;
    sched_fence();   // bound register pressure in precompute
  }

  // ---- gi_ctx = context @ W_ih[:,2:].T + b_ih (+ b_hh folded for r,z gates) ----
#pragma unroll 1
  for (int nt = 0; nt < 24; nt++) {
    v8f acc = {};
    const float* wrow = W_ih + (size_t)(nt * 16 + col) * 258 + 2;
#pragma unroll
    for (int kt = 0; kt < 8; kt++) {
      const v2f* p = (const v2f*)(wrow + kt * 32 + hi * 16);  // 8B aligned
      v16h b;
#pragma unroll
      for (int u = 0; u < 8; u++) {
        v2f q = p[u];
        b[u * 2 + 0] = (f16)q[0];
        b[u * 2 + 1] = (f16)q[1];
      }
      acc = wmma_f16(actx[kt], b, acc);
    }
    int j = nt * 16 + col;
    float bias = b_ih[j] + (nt < 16 ? b_hh[j] : 0.0f);  // fold r/z recurrent bias
    v8h g;
#pragma unroll
    for (int r = 0; r < 8; r++) g[r] = (f16)(acc[r] + bias);
    *(v8h*)&sGi[wv][nt * 256 + col * 16 + hi * 8] = g;
  }

  // ---- hoist time-invariant per-lane scalars ----
  float bhn[8], bb1[4];
#pragma unroll
  for (int c = 0; c < 8; c++) bhn[c] = b_hh[(c + 16) * 16 + col];
#pragma unroll
  for (int c2 = 0; c2 < 4; c2++) bb1[c2] = b1[c2 * 16 + col];
  const float b2v = (col < 2) ? b2[col] : 0.0f;
  // output base for this lane (stores become base + r*384B immediate offsets)
  float* orow = out + ((size_t)(rowBase + hi * 8) * TSTEPS) * 2 + col;

  // ---- main recurrence: 48 sequential GRU steps ----
  v8f pos = {};
#pragma unroll 1
  for (int t = 0; t < TSTEPS; t++) {
    // broadcast previous delta per row (same-wave LDS round-trip)
    float d0[8], d1[8];
#pragma unroll
    for (int r = 0; r < 8; r++) {
      int m = hi * 8 + r;
      d0[r] = sDelta[wv][m * 2 + 0];
      d1[r] = sDelta[wv][m * 2 + 1];
    }
    // A-fragments of h (f16 mirror in LDS)
    v16h ah[4];
#pragma unroll
    for (int kt = 0; kt < 4; kt++) {
      v8h lo = *(const v8h*)&sH[wv][col * 128 + kt * 32 + hi * 16];
      v8h hh = *(const v8h*)&sH[wv][col * 128 + kt * 32 + hi * 16 + 8];
      ah[kt] = cat16(lo, hh);
    }
    // gate tiles: fully unrolled (constant offsets), fenced per iteration
#pragma unroll
    for (int c = 0; c < 8; c++) {
      v8f gr = {}, gz = {}, gn = {};
#pragma unroll
      for (int kt = 0; kt < 4; kt++) gr = wmma_f16(ah[kt], ldsB(sWhh, (c      ) * 4 + kt, ln), gr);
#pragma unroll
      for (int kt = 0; kt < 4; kt++) gz = wmma_f16(ah[kt], ldsB(sWhh, (c +  8) * 4 + kt, ln), gz);
#pragma unroll
      for (int kt = 0; kt < 4; kt++) gn = wmma_f16(ah[kt], ldsB(sWhh, (c + 16) * 4 + kt, ln), gn);
      v8h gir = *(const v8h*)&sGi[wv][(c      ) * 256 + col * 16 + hi * 8];
      v8h giz = *(const v8h*)&sGi[wv][(c +  8) * 256 + col * 16 + hi * 8];
      v8h gin = *(const v8h*)&sGi[wv][(c + 16) * 256 + col * 16 + hi * 8];
      v2f wr = *(const v2f*)&sWd[((c      ) * 16 + col) * 2];
      v2f wz = *(const v2f*)&sWd[((c +  8) * 16 + col) * 2];
      v2f wn = *(const v2f*)&sWd[((c + 16) * 16 + col) * 2];
      v8f hv = hst[c];
#pragma unroll
      for (int r = 0; r < 8; r++) {
        float ir  = (float)gir[r] + d0[r] * wr[0] + d1[r] * wr[1] + gr[r];
        float iz  = (float)giz[r] + d0[r] * wz[0] + d1[r] * wz[1] + gz[r];
        float in_ = (float)gin[r] + d0[r] * wn[0] + d1[r] * wn[1];
        float rr = sigm(ir);
        float zz = sigm(iz);
        float nn = tanh_fast(in_ + rr * (gn[r] + bhn[c]));
        float hnew = (1.0f - zz) * nn + zz * hv[r];
        hv[r] = hnew;
        sH[wv][(hi * 8 + r) * 128 + c * 16 + col] = (f16)hnew;
      }
      hst[c] = hv;
      sched_fence();   // cap cross-iteration hoisting -> keep VGPRs < 256
    }
    // MLP layer 1: hid = relu(h_new @ W1.T + b1)
    v16h ah2[4];
#pragma unroll
    for (int kt = 0; kt < 4; kt++) {
      v8h lo = *(const v8h*)&sH[wv][col * 128 + kt * 32 + hi * 16];
      v8h hh = *(const v8h*)&sH[wv][col * 128 + kt * 32 + hi * 16 + 8];
      ah2[kt] = cat16(lo, hh);
    }
#pragma unroll
    for (int c2 = 0; c2 < 4; c2++) {
      v8f acc = {};
#pragma unroll
      for (int kt = 0; kt < 4; kt++) acc = wmma_f16(ah2[kt], ldsB(sW1, c2 * 4 + kt, ln), acc);
#pragma unroll
      for (int r = 0; r < 8; r++) {
        float v = acc[r] + bb1[c2];
        v = v > 0.0f ? v : 0.0f;
        sHid[wv][(hi * 8 + r) * 64 + c2 * 16 + col] = (f16)v;
      }
    }
    sched_fence();
    // MLP layer 2: pred_delta = hid @ W2.T + b2  (N padded to 16)
    v8f pd = {};
#pragma unroll
    for (int kt = 0; kt < 2; kt++) {
      v8h lo = *(const v8h*)&sHid[wv][col * 64 + kt * 32 + hi * 16];
      v8h hh = *(const v8h*)&sHid[wv][col * 64 + kt * 32 + hi * 16 + 8];
      pd = wmma_f16(cat16(lo, hh), ldsB(sW2, kt, ln), pd);
    }
#pragma unroll
    for (int r = 0; r < 8; r++) pos[r] += pd[r] + b2v;
    // write positions + feed delta back (base + constant offsets)
    if (col < 2) {
#pragma unroll
      for (int r = 0; r < 8; r++) {
        int m = hi * 8 + r;
        orow[(size_t)r * (TSTEPS * 2) + t * 2] = pos[r];
        sDelta[wv][m * 2 + col] = pd[r] + b2v;
      }
    }
  }
}

extern "C" void kernel_launch(void* const* d_in, const int* in_sizes, int n_in,
                              void* d_out, int out_size, void* d_ws, size_t ws_size,
                              hipStream_t stream) {
  (void)in_sizes; (void)n_in; (void)d_ws; (void)ws_size; (void)out_size;
  const float* ctx    = (const float*)d_in[0];
  const float* lv     = (const float*)d_in[1];
  const float* W_init = (const float*)d_in[2];
  const float* b_init = (const float*)d_in[3];
  const float* W_ih   = (const float*)d_in[4];
  const float* b_ih   = (const float*)d_in[5];
  const float* W_hh   = (const float*)d_in[6];
  const float* b_hh   = (const float*)d_in[7];
  const float* W1     = (const float*)d_in[8];
  const float* b1     = (const float*)d_in[9];
  const float* W2     = (const float*)d_in[10];
  const float* b2     = (const float*)d_in[11];
  float* out = (float*)d_out;
  dim3 grid(BATCH / 32);   // 512 workgroups, 2 waves * 16 rows each
  dim3 block(64);
  delta_decoder_gru_kernel<<<grid, block, 0, stream>>>(
      ctx, lv, W_init, b_init, W_ih, b_ih, W_hh, b_hh, W1, b1, W2, b2, out);
}
